// ResNet_imagenet_3719441678861
// MI455X (gfx1250) — 8-bit matrix operand order repaired and checked by reading, compiled, not run on hardware
//
#include <hip/hip_runtime.h>
#include <cstdint>
#include <cstddef>

typedef int v8i __attribute__((ext_vector_type(8)));

__global__ void conv1_direct(const float* __restrict__ x,
                             const float* __restrict__ w,
                             float* __restrict__ out)
{
    int idx = blockIdx.x * blockDim.x + threadIdx.x;
    const int total = 16 * 112 * 112 * 64;
    if (idx >= total) return;
    int co = idx & 63;
    int t  = idx >> 6;
    int wo = t % 112; t /= 112;
    int ho = t % 112;
    int n  = t / 112;
    float acc = 0.f;
    for (int ci = 0; ci < 3; ++ci) {
        const float* xp = x + ((size_t)(n * 3 + ci)) * 224 * 224;
        const float* wp = w + ((size_t)(co * 3 + ci)) * 49;
        for (int kh = 0; kh < 7; ++kh) {
            int h = ho * 2 - 3 + kh;
            if ((unsigned)h >= 224u) continue;
            for (int kw = 0; kw < 7; ++kw) {
                int ww = wo * 2 - 3 + kw;
                if ((unsigned)ww >= 224u) continue;
                float wv = wp[kh * 7 + kw];
                float xv = xp[h * 224 + ww];
                acc += (wv > 0.f) ? xv : ((wv < 0.f) ? -xv : 0.f);
            }
        }
    }
    out[idx] = acc;
}

__global__ void maxpool3s2(const float* __restrict__ in, float* __restrict__ out,
                           int Hin, int Win, int Hout, int Wout, int C, int total)
{
    int idx = blockIdx.x * blockDim.x + threadIdx.x;
    if (idx >= total) return;
    int c = idx % C; int t = idx / C;
    int wo = t % Wout; t /= Wout;
    int ho = t % Hout;
    int n  = t / Hout;
    float m = -3.402823466e38f;
    for (int kh = 0; kh < 3; ++kh) {
        int h = ho * 2 - 1 + kh;
        if ((unsigned)h >= (unsigned)Hin) continue;
        for (int kw = 0; kw < 3; ++kw) {
            int ww = wo * 2 - 1 + kw;
            if ((unsigned)ww >= (unsigned)Win) continue;
            float v = in[((size_t)(n * Hin + h) * Win + ww) * C + c];
            m = fmaxf(m, v);
        }
    }
    out[idx] = m;
}

__global__ void pack_sign(const float* __restrict__ in, int8_t* __restrict__ out, int total)
{
    int idx = blockIdx.x * blockDim.x + threadIdx.x;
    if (idx >= total) return;
    float v = in[idx];
    out[idx] = (v > 0.f) ? (int8_t)1 : ((v < 0.f) ? (int8_t)-1 : (int8_t)0);
}

__global__ void pack_weights(const float* __restrict__ w, int* __restrict__ Bp,
                             int C, int log2C, int KH, int KW,
                             int CoutP, int CoutReal, int K)
{
    int idx = blockIdx.x * blockDim.x + threadIdx.x;
    int total = (K >> 2) * CoutP;
    if (idx >= total) return;
    int co = idx % CoutP;
    int q  = idx / CoutP;
    unsigned pack = 0u;
    if (co < CoutReal) {
        for (int b = 0; b < 4; ++b) {
            int k  = q * 4 + b;
            int seg = k >> log2C;
            int ci  = k & (C - 1);
            int kh  = seg / KW;
            int kw  = seg - kh * KW;
            float v = w[(((size_t)co * C + ci) * KH + kh) * KW + kw];
            int8_t s = (v > 0.f) ? (int8_t)1 : ((v < 0.f) ? (int8_t)-1 : (int8_t)0);
            pack |= ((unsigned)(uint8_t)s) << (8 * b);
        }
    }
    Bp[(size_t)q * CoutP + co] = (int)pack;
}

__global__ void binconv_wmma(const int8_t* __restrict__ S, const int* __restrict__ Bp,
                             int* __restrict__ out,
                             int H, int W, int C, int log2C,
                             int KH, int KW, int stride, int pad,
                             int Ho, int Wo, int Cout, int M, int K)
{
    int lane = threadIdx.x & 31;
    int wave = threadIdx.x >> 5;
    int m0 = (blockIdx.x * (blockDim.x >> 5) + wave) * 16;
    if (m0 >= M) return;
    int n0 = blockIdx.y * 64;
    int hf = lane >> 4;
    int r  = lane & 15;

    int m    = m0 + r;
    int hw   = Ho * Wo;
    int nimg = m / hw;
    int rem  = m - nimg * hw;
    int ho   = rem / Wo;
    int wo   = rem - ho * Wo;
    int hb   = ho * stride - pad;
    int wb   = wo * stride - pad;
    const int8_t* Sn = S + (size_t)nimg * H * W * C;
    int ncol = n0 + r;
    int Cm1  = C - 1;

    int ak[8], boff[8];
#pragma unroll
    for (int i = 0; i < 8; ++i) {
        int g = i >> 2, j = i & 3;
        ak[i]   = g * 32 + hf * 16 + j * 4;
        boff[i] = g * 8 + hf * 4 + j;
    }

    v8i acc[4] = {};
    for (int kstep = 0; kstep < K; kstep += 64) {
        int seg = kstep >> log2C;
        int ci0 = kstep & Cm1;
        int kh  = seg / KW;
        int kw  = seg - kh * KW;
        int h = hb + kh, ww = wb + kw;
        bool inb = ((unsigned)h < (unsigned)H) && ((unsigned)ww < (unsigned)W);
        const int8_t* ap = Sn + ((size_t)(h * W + ww)) * C + ci0;

        v8i a;
#pragma unroll
        for (int i = 0; i < 8; ++i)
            a[i] = inb ? *(const int*)(ap + ak[i]) : 0;

        int q0 = kstep >> 2;
        if (kstep + 64 < K)
            __builtin_prefetch(Bp + (size_t)(q0 + 16) * Cout + ncol, 0, 0);

#pragma unroll
        for (int t = 0; t < 4; ++t) {
            v8i b;
#pragma unroll
            for (int i = 0; i < 8; ++i)
                b[i] = Bp[(size_t)(q0 + boff[i]) * Cout + ncol + t * 16];
            acc[t] = __builtin_amdgcn_wmma_i32_16x16x64_iu8(true, a, true, b, acc[t],
                                                            false, false);
        }
    }

#pragma unroll
    for (int t = 0; t < 4; ++t) {
#pragma unroll
        for (int i = 0; i < 8; ++i) {
            int row = hf * 8 + i;
            out[(size_t)(m0 + row) * Cout + n0 + t * 16 + r] = acc[t][i];
        }
    }
}

template <typename T>
__global__ void bn_stats(const T* __restrict__ in, float* __restrict__ stats,
                         int M, int Cout)
{
    int c = blockIdx.y * blockDim.x + threadIdx.x;
    float s = 0.f, s2 = 0.f;
    for (int m = blockIdx.x; m < M; m += gridDim.x) {
        float v = (float)in[(size_t)m * Cout + c];
        s += v; s2 += v * v;
    }
    atomicAdd(&stats[c], s);
    atomicAdd(&stats[Cout + c], s2);
}

template <typename T>
__global__ void bn_apply(const T* __restrict__ in, const float* __restrict__ stats,
                         const float* __restrict__ gamma, const float* __restrict__ beta,
                         const float* __restrict__ residual,
                         float* __restrict__ outF, int8_t* __restrict__ outS,
                         int M, int Cout, int clampf, float invM)
{
    size_t total = (size_t)M * Cout;
    size_t idx = (size_t)blockIdx.x * blockDim.x + threadIdx.x;
    if (idx >= total) return;
    int c = (int)(idx % Cout);
    float mean = stats[c] * invM;
    float var  = stats[Cout + c] * invM - mean * mean;
    float sc = gamma[c] * rsqrtf(var + 1e-5f);
    float sh = beta[c] - mean * sc;
    float v = (float)in[idx] * sc + sh;
    if (residual) v += residual[idx];
    if (clampf)   v = fminf(1.f, fmaxf(-1.f, v));
    if (outF) outF[idx] = v;
    if (outS) outS[idx] = (v > 0.f) ? (int8_t)1 : ((v < 0.f) ? (int8_t)-1 : (int8_t)0);
}

__global__ void avgpool_sign(const float* __restrict__ in, int8_t* __restrict__ outS)
{
    int idx = blockIdx.x * blockDim.x + threadIdx.x;
    if (idx >= 16 * 2048) return;
    int c = idx & 2047;
    int n = idx >> 11;
    float s = 0.f;
    for (int p = 0; p < 49; ++p)
        s += in[((size_t)n * 49 + p) * 2048 + c];
    float v = s * (1.f / 49.f);
    outS[idx] = (v > 0.f) ? (int8_t)1 : ((v < 0.f) ? (int8_t)-1 : (int8_t)0);
}

__global__ void fc_wmma_kernel(const int8_t* __restrict__ A, const int* __restrict__ Bp,
                               const float* __restrict__ bias, float* __restrict__ out,
                               int K, int CoutP, int Nreal)
{
    int lane = threadIdx.x & 31;
    int hf = lane >> 4;
    int r  = lane & 15;
    int n0 = blockIdx.x * 16;
    int ncol = n0 + r;
    int ak[8], boff[8];
#pragma unroll
    for (int i = 0; i < 8; ++i) {
        int g = i >> 2, j = i & 3;
        ak[i]   = g * 32 + hf * 16 + j * 4;
        boff[i] = g * 8 + hf * 4 + j;
    }
    v8i acc = {};
    for (int kstep = 0; kstep < K; kstep += 64) {
        int q0 = kstep >> 2;
        v8i a, b;
#pragma unroll
        for (int i = 0; i < 8; ++i) {
            a[i] = *(const int*)(A + (size_t)r * K + kstep + ak[i]);
            b[i] = Bp[(size_t)(q0 + boff[i]) * CoutP + ncol];
        }
        acc = __builtin_amdgcn_wmma_i32_16x16x64_iu8(true, a, true, b, acc, false, false);
    }
    if (ncol < Nreal) {
        float bi = bias[ncol];
#pragma unroll
        for (int i = 0; i < 8; ++i) {
            int row = hf * 8 + i;
            out[(size_t)row * Nreal + ncol] = (float)acc[i] + bi;
        }
    }
}

static void bin_conv_bn(hipStream_t st, const int8_t* Sin, int H, int W, int C,
                        const float* wOIHW, int KH, int KW, int stride, int pad, int Cout,
                        const float* g, const float* b,
                        const float* res, float* outF, int8_t* outS, int clampf,
                        int* BP, int* I32, float* STATS)
{
    int K  = C * KH * KW;
    int Ho = (H + 2 * pad - KH) / stride + 1;
    int Wo = (W + 2 * pad - KW) / stride + 1;
    int M  = 16 * Ho * Wo;
    int log2C = 31 - __builtin_clz((unsigned)C);

    int totW = (K >> 2) * Cout;
    pack_weights<<<(totW + 255) / 256, 256, 0, st>>>(wOIHW, BP, C, log2C, KH, KW, Cout, Cout, K);

    dim3 grid((unsigned)((M + 63) / 64), (unsigned)(Cout / 64));
    binconv_wmma<<<grid, 128, 0, st>>>(Sin, BP, I32, H, W, C, log2C,
                                       KH, KW, stride, pad, Ho, Wo, Cout, M, K);

    hipMemsetAsync(STATS, 0, 2 * (size_t)Cout * sizeof(float), st);
    bn_stats<int><<<dim3(256, (unsigned)(Cout / 64)), 64, 0, st>>>(I32, STATS, M, Cout);

    size_t tot = (size_t)M * Cout;
    bn_apply<int><<<(unsigned)((tot + 255) / 256), 256, 0, st>>>(
        I32, STATS, g, b, res, outF, outS, M, Cout, clampf, 1.0f / (float)M);
}

extern "C" void kernel_launch(void* const* d_in, const int* in_sizes, int n_in,
                              void* d_out, int out_size, void* d_ws, size_t ws_size,
                              hipStream_t stream)
{
    (void)in_sizes; (void)n_in; (void)out_size; (void)ws_size;
    static const int LAYERS[4]  = {3, 4, 6, 3};
    static const int PLANES[4]  = {64, 128, 256, 512};
    static const int STRIDES[4] = {1, 2, 2, 2};

    int cur = 0;
    auto nxt = [&]() -> const float* { return (const float*)d_in[cur++]; };

    const float* x       = nxt();
    const float* conv1_w = nxt();
    const float* bn1_g   = nxt();
    const float* bn1_b   = nxt();

    const size_t EL = 16ull * 112 * 112 * 64;
    char* ws = (char*)d_ws;
    float*  X    = (float*)(ws);
    float*  Y    = (float*)(ws + EL * 4);
    float*  RES  = (float*)(ws + EL * 8);
    int*    I32  = (int*)  (ws + EL * 12);
    int8_t* S    = (int8_t*)(ws + EL * 16);
    int*    BP   = (int*)  (ws + EL * 17);
    float*  STATS= (float*)(ws + EL * 17 + (4u << 20));
    int8_t* PS   = (int8_t*)(ws + EL * 17 + (4u << 20) + (64u << 10));

    float* C1 = (float*)I32;
    {
        int total = 16 * 112 * 112 * 64;
        conv1_direct<<<(total + 255) / 256, 256, 0, stream>>>(x, conv1_w, C1);
        int M1 = 16 * 112 * 112;
        hipMemsetAsync(STATS, 0, 2 * 64 * sizeof(float), stream);
        bn_stats<float><<<dim3(256, 1), 64, 0, stream>>>(C1, STATS, M1, 64);
        size_t tot = (size_t)M1 * 64;
        bn_apply<float><<<(unsigned)((tot + 255) / 256), 256, 0, stream>>>(
            C1, STATS, bn1_g, bn1_b, nullptr, Y, nullptr, M1, 64, 1, 1.0f / (float)M1);
        int totp = 16 * 56 * 56 * 64;
        maxpool3s2<<<(totp + 255) / 256, 256, 0, stream>>>(Y, X, 112, 112, 56, 56, 64, totp);
        pack_sign<<<(totp + 255) / 256, 256, 0, stream>>>(X, S, totp);
    }

    int H = 56, W = 56, Cin = 64;
    for (int s = 0; s < 4; ++s) {
        int planes = PLANES[s];
        for (int i = 0; i < LAYERS[s]; ++i) {
            int stride = (i == 0) ? STRIDES[s] : 1;
            const float* c1w = nxt(); const float* b1g = nxt(); const float* b1b = nxt();
            const float* c2w = nxt(); const float* b2g = nxt(); const float* b2b = nxt();
            const float* c3w = nxt(); const float* b3g = nxt(); const float* b3b = nxt();
            const float* dw = nullptr; const float* dg = nullptr; const float* db = nullptr;
            bool down = (i == 0);
            if (down) { dw = nxt(); dg = nxt(); db = nxt(); }

            int Ho = (H + 2 - 3) / stride + 1;
            int Wo = (W + 2 - 3) / stride + 1;

            const float* resPtr;
            if (down) {
                bin_conv_bn(stream, S, H, W, Cin, dw, 1, 1, stride, 0, 4 * planes,
                            dg, db, nullptr, RES, nullptr, 0, BP, I32, STATS);
                resPtr = RES;
            } else {
                resPtr = X;
            }
            bin_conv_bn(stream, S, H, W, Cin, c1w, 1, 1, 1, 0, planes,
                        b1g, b1b, nullptr, nullptr, S, 1, BP, I32, STATS);
            bin_conv_bn(stream, S, H, W, planes, c2w, 3, 3, stride, 1, planes,
                        b2g, b2b, nullptr, nullptr, S, 1, BP, I32, STATS);
            bin_conv_bn(stream, S, Ho, Wo, planes, c3w, 1, 1, 1, 0, 4 * planes,
                        b3g, b3b, resPtr, Y, S, 1, BP, I32, STATS);

            float* t = X; X = Y; Y = t;
            H = Ho; W = Wo; Cin = 4 * planes;
        }
    }

    const float* fc_w = nxt();
    const float* fc_b = nxt();

    avgpool_sign<<<(16 * 2048 + 255) / 256, 256, 0, stream>>>(X, PS);
    {
        const int K = 2048, CoutP = 1008, Nreal = 1000;
        int totW = (K >> 2) * CoutP;
        pack_weights<<<(totW + 255) / 256, 256, 0, stream>>>(fc_w, BP, K, 11, 1, 1,
                                                             CoutP, Nreal, K);
        fc_wmma_kernel<<<CoutP / 16, 32, 0, stream>>>(PS, BP, fc_b, (float*)d_out,
                                                      K, CoutP, Nreal);
    }
}
